// Attention_8289286881814
// MI455X (gfx1250) — matrix operand order repaired and checked by reading, not compiled, not run on hardware
//
#include <hip/hip_runtime.h>
#include <hip/hip_bf16.h>

#define B_    32
#define T_IN  512
#define D_    256
#define T_OUT 4096
#define EPS   1e-6f
#define TM    64          // output frames per block
#define W_STRIDE 520      // halves per LDS weight row: 16B-aligned, stride/4 dwords = 260 (4*odd)

typedef __attribute__((ext_vector_type(16))) _Float16 v16h;
typedef __attribute__((ext_vector_type(8)))  _Float16 v8h;
typedef __attribute__((ext_vector_type(8)))  float    v8f;

// ---------------------------------------------------------------------------
// Kernel 1: per-batch inclusive scan of durations -> centers; 1/r^2
// ---------------------------------------------------------------------------
__global__ __launch_bounds__(T_IN)
void scan_centers_kernel(const float* __restrict__ dur, const float* __restrict__ rng,
                         float* __restrict__ c_g, float* __restrict__ ir_g) {
    __shared__ float s[T_IN];
    const int b = blockIdx.x, tid = threadIdx.x;
    const float d = dur[b * T_IN + tid];
    s[tid] = d;
    __syncthreads();
    for (int off = 1; off < T_IN; off <<= 1) {
        float v = (tid >= off) ? s[tid - off] : 0.0f;
        __syncthreads();
        s[tid] += v;
        __syncthreads();
    }
    c_g[b * T_IN + tid] = s[tid] - 0.5f * d;
    const float r = rng[b * T_IN + tid];
    ir_g[b * T_IN + tid] = 1.0f / (r * r);
}

// ---------------------------------------------------------------------------
// Kernel 2: transpose + convert embeddings f32 (B,T_IN,D) -> f16 (B,D,T_IN)
// so WMMA B-fragments are contiguous 32B loads per lane.
// ---------------------------------------------------------------------------
__global__ __launch_bounds__(256)
void transpose_f16_kernel(const float* __restrict__ emb, _Float16* __restrict__ embT) {
    __shared__ float tile[32][33];
    const int t0 = blockIdx.x * 32, d0 = blockIdx.y * 32, b = blockIdx.z;
    const int tx = threadIdx.x, ty = threadIdx.y;   // block (32, 8)
    for (int r = ty; r < 32; r += 8)
        tile[r][tx] = emb[(size_t)(b * T_IN + t0 + r) * D_ + d0 + tx];
    __syncthreads();
    for (int r = ty; r < 32; r += 8)
        embT[(size_t)(b * D_ + d0 + r) * T_IN + t0 + tx] = (_Float16)tile[tx][r];
}

// ---------------------------------------------------------------------------
// Kernel 3: fused weights -> WMMA matmul -> normalize -> +PE -> store
// grid (T_OUT/TM, B), block 256 (8 waves of 32)
// ---------------------------------------------------------------------------
__global__ __launch_bounds__(256)
void upsample_wmma_kernel(const _Float16* __restrict__ embT,
                          const float* __restrict__ c_g,
                          const float* __restrict__ ir_g,
                          float* __restrict__ out) {
    extern __shared__ char smem_raw[];
    _Float16* w_lds = (_Float16*)smem_raw;                       // TM * W_STRIDE halves
    float* c_l  = (float*)(smem_raw + TM * W_STRIDE * 2);        // T_IN
    float* ir_l = c_l + T_IN;                                    // T_IN
    float* part = ir_l + T_IN;                                   // 4 * TM
    float* invs = part + 4 * TM;                                 // TM

    const int tid = threadIdx.x;
    const int b   = blockIdx.y;
    const int t0  = blockIdx.x * TM;

    for (int i = tid; i < T_IN; i += 256) {
        c_l[i]  = c_g[b * T_IN + i];
        ir_l[i] = ir_g[b * T_IN + i];
    }
    __syncthreads();

    // ---- unnormalized Gaussian weights into LDS (f16), track column sums ----
    const int tcol = tid & (TM - 1);
    const int igrp = tid >> 6;                 // 0..3
    const float tval = (float)(t0 + tcol);
    float sum = 0.0f;
    _Float16* wrow = w_lds + tcol * W_STRIDE;
    for (int i = igrp; i < T_IN; i += 4) {
        const float dt = tval - c_l[i];
        const float w  = __expf(-ir_l[i] * dt * dt);
        sum += w;
        wrow[i] = (_Float16)w;
    }
    part[igrp * TM + tcol] = sum;
    __syncthreads();
    if (tid < TM) {
        const float s = part[tid] + part[TM + tid] + part[2 * TM + tid] + part[3 * TM + tid];
        invs[tid] = 1.0f / (s + EPS);
    }
    __syncthreads();

    // ---- WMMA: wave -> 1 M-tile x 8 N-tiles, K loop over 512 in steps of 32 ----
    const int wid   = tid >> 5;
    const unsigned lane = tid & 31u;
    const int mt    = wid >> 1;                // 0..3   (M tile)
    const int nt0   = (wid & 1) * 8;           // tile col base (each tile = 16 d)
    const int lhalf = lane >> 4;               // 0/1
    const int l16   = lane & 15;

    const _Float16* eb = embT + (size_t)b * D_ * T_IN;
    const _Float16* arow = w_lds + (mt * 16 + l16) * W_STRIDE;

    v8f acc[8] = {};
    for (int k0 = 0; k0 < T_IN; k0 += 32) {
        // A fragment (contiguous K order, matches B): lanes 0-15: K=0-15; lanes 16-31: K=16-31
        const int ka = k0 + lhalf * 16;
        const v8h lo = *(const v8h*)(arow + ka);
        const v8h hi = *(const v8h*)(arow + ka + 8);
        const v16h a = __builtin_shufflevector(lo, hi,
            0, 1, 2, 3, 4, 5, 6, 7, 8, 9, 10, 11, 12, 13, 14, 15);
#pragma unroll
        for (int j = 0; j < 8; ++j) {
            // B fragment: N = l16, 16 consecutive K per lane from transposed f16 embeddings
            const int dcol = (nt0 + j) * 16 + l16;
            const v16h bm = *(const v16h*)(eb + (size_t)dcol * T_IN + k0 + lhalf * 16);
            acc[j] = __builtin_amdgcn_wmma_f32_16x16x32_f16(
                false, a, false, bm, (short)0, acc[j], false, false);
        }
    }

    // ---- epilogue: per-row normalization, sinusoidal PE, store ----
    const float LOG_DIV = -2.0f * 9.210340371976184f / (float)D_;   // -2*ln(10000)/D
#pragma unroll
    for (int j = 0; j < 8; ++j) {
        const int dcol  = (nt0 + j) * 16 + l16;
        const float dv  = __expf(LOG_DIV * (float)(dcol >> 1));
        const bool oddd = (dcol & 1) != 0;
#pragma unroll
        for (int r = 0; r < 8; ++r) {
            const int m = mt * 16 + r + lhalf * 8;        // C-layout: lanes>=16 hold M=r+8
            float v = acc[j][r] * invs[m];
            const float ang = (float)(t0 + m) * dv;
            v += oddd ? __cosf(ang) : __sinf(ang);
            out[((size_t)b * T_OUT + (t0 + m)) * D_ + dcol] = v;
        }
    }
}

// ---------------------------------------------------------------------------
extern "C" void kernel_launch(void* const* d_in, const int* in_sizes, int n_in,
                              void* d_out, int out_size, void* d_ws, size_t ws_size,
                              hipStream_t stream) {
    const float* emb = (const float*)d_in[0];
    const float* dur = (const float*)d_in[1];
    const float* rng = (const float*)d_in[2];
    float* out = (float*)d_out;

    char* ws = (char*)d_ws;
    float*    c_g  = (float*)ws;                                   // B*T_IN f32
    float*    ir_g = (float*)(ws + (size_t)B_ * T_IN * 4);         // B*T_IN f32
    _Float16* embT = (_Float16*)(ws + (size_t)2 * B_ * T_IN * 4);  // B*D*T_IN f16 (~8 MB)

    scan_centers_kernel<<<B_, T_IN, 0, stream>>>(dur, rng, c_g, ir_g);
    transpose_f16_kernel<<<dim3(T_IN / 32, D_ / 32, B_), dim3(32, 8), 0, stream>>>(emb, embT);

    const size_t shmem = (size_t)TM * W_STRIDE * 2   // weights f16
                       + (size_t)T_IN * 4 * 2        // c, ir
                       + (size_t)4 * TM * 4          // partial sums
                       + (size_t)TM * 4;             // inv sums
    upsample_wmma_kernel<<<dim3(T_OUT / TM, B_), 256, shmem, stream>>>(embT, c_g, ir_g, out);
}
